// EGNN_Full_29076928594665
// MI455X (gfx1250) — compile-verified, matrix operand order checked by reading, not run on hardware
//
#include <hip/hip_runtime.h>
#include <hip/hip_bf16.h>

// ---------------------------------------------------------------------------
// EGNN fused implementation for gfx1250 (CDNA5, wave32, WMMA)
// 32 rows/block, dual-accumulator waves: each B fragment feeds 2 WMMAs.
// ---------------------------------------------------------------------------

typedef __attribute__((ext_vector_type(16))) _Float16 v16h;
typedef __attribute__((ext_vector_type(8)))  float    v8f;

namespace {

constexpr int N_NODES = 50000;
constexpr int N_EDGES = 800000;
constexpr int H       = 128;
constexpr int DEPTH   = 5;
constexpr int G       = 64;

constexpr int KM1 = 288;   // 2H+1 = 257 padded to multiple of 32
constexpr int KU1 = 256;   // 2H
constexpr int K2  = 128;   // H

constexpr int TM = 32;     // rows (edges/nodes) per block

constexpr float EPS = 1e-5f;

// -------------------- fragment-swizzled weight packing ---------------------
// Packed layout: out[(((kt*8 + nt)*32 + lane)*16 + slot)]
// k_local = (v<4 ? 2v : 16+2(v-4)) + p + 8*group   (CDNA5 16-bit WMMA layout)
__global__ __launch_bounds__(256) void pack_w_kernel(
    const float* __restrict__ W, int Korig, int ktiles, _Float16* __restrict__ out)
{
    int idx = blockIdx.x * blockDim.x + threadIdx.x;
    int total = ktiles * 8 * 32 * 16;
    if (idx >= total) return;
    int slot = idx & 15;
    int lane = (idx >> 4) & 31;
    int nt   = (idx >> 9) & 7;
    int kt   = idx >> 12;
    int grp  = lane >> 4;
    int nlo  = lane & 15;
    int v    = slot >> 1;
    int p    = slot & 1;
    int kl   = ((v < 4) ? (2 * v) : (16 + 2 * (v - 4))) + p + 8 * grp;
    int k    = kt * 32 + kl;
    int n    = nt * 16 + nlo;
    float val = (k < Korig) ? W[(size_t)k * H + n] : 0.0f;
    out[idx] = (_Float16)val;
}

// ---- per-wave dual 16x16 WMMA tiles (M rows 0-15 and 16-31) over K --------
__device__ __forceinline__ void wmma_tile2(const _Float16* __restrict__ A, int ldA,
                                           int ktiles,
                                           const _Float16* __restrict__ Bpack,
                                           int nt, int lane,
                                           v8f& acc0, v8f& acc1)
{
    acc0 = {};
    acc1 = {};
    int m   = lane & 15;
    int grp = lane >> 4;
    for (int kt = 0; kt < ktiles; ++kt) {
        union { v16h v; unsigned u[8]; } a0, a1, b;
        // B fragment: contiguous 32B per lane (pre-swizzled), reused twice
        const unsigned* bp =
            (const unsigned*)(Bpack + ((((size_t)kt * 8 + nt) * 32 + lane) * 16));
#pragma unroll
        for (int j = 0; j < 8; ++j) b.u[j] = bp[j];
        // A fragments (rows m and 16+m) from LDS at swizzled K offsets
        const unsigned* p00 =
            (const unsigned*)(A + (size_t)m * ldA + kt * 32 + grp * 8);
        const unsigned* p01 =
            (const unsigned*)(A + (size_t)m * ldA + kt * 32 + grp * 8 + 16);
        const unsigned* p10 =
            (const unsigned*)(A + (size_t)(16 + m) * ldA + kt * 32 + grp * 8);
        const unsigned* p11 =
            (const unsigned*)(A + (size_t)(16 + m) * ldA + kt * 32 + grp * 8 + 16);
#pragma unroll
        for (int j = 0; j < 4; ++j) {
            a0.u[j] = p00[j]; a0.u[4 + j] = p01[j];
            a1.u[j] = p10[j]; a1.u[4 + j] = p11[j];
        }
        acc0 = __builtin_amdgcn_wmma_f32_16x16x32_f16(
            false, a0.v, false, b.v, (short)0, acc0, false, false);
        acc1 = __builtin_amdgcn_wmma_f32_16x16x32_f16(
            false, a1.v, false, b.v, (short)0, acc1, false, false);
    }
}

// store two D tiles (+bias) into St[32][128]
__device__ __forceinline__ void store_tiles(float* __restrict__ St,
                                            const v8f& acc0, const v8f& acc1,
                                            const float* __restrict__ bias,
                                            int nt, int lane)
{
    int m = lane & 15, grp = lane >> 4;
    int col = nt * 16 + m;
    float bb = bias[col];
#pragma unroll
    for (int r = 0; r < 8; ++r) {
        St[(r + 8 * grp) * H + col]      = acc0[r] + bb;
        St[(16 + r + 8 * grp) * H + col] = acc1[r] + bb;
    }
}

// -------------------- LayerNorm statistics over 32x128 tile ----------------
__device__ __forceinline__ void ln_stats(const float* __restrict__ St, int tid,
                                         float* __restrict__ red,
                                         float* __restrict__ red2,
                                         float* __restrict__ mu,
                                         float* __restrict__ rs)
{
    {   // 256 threads = 32 rows x 8 segments
        int r = tid >> 3, s = tid & 7;
        float sm = 0.f, sq = 0.f;
        const float* p = St + r * H + s * 16;
#pragma unroll
        for (int j = 0; j < 16; ++j) { float v = p[j]; sm += v; sq += v * v; }
        red[tid]  = sm;
        red2[tid] = sq;
    }
    __syncthreads();
    if (tid < TM) {
        float sm = 0.f, sq = 0.f;
#pragma unroll
        for (int j = 0; j < 8; ++j) { sm += red[tid * 8 + j]; sq += red2[tid * 8 + j]; }
        float m   = sm * (1.0f / 128.0f);
        float var = sq * (1.0f / 128.0f) - m * m;
        mu[tid] = m;
        rs[tid] = rsqrtf(var + EPS);
    }
    __syncthreads();
}

// -------------------- embedding / dist / pooling ---------------------------
__global__ __launch_bounds__(256) void emb_kernel(
    const float* __restrict__ x, const float* __restrict__ ew,
    const float* __restrict__ eb, float* __restrict__ h)
{
    int idx = blockIdx.x * blockDim.x + threadIdx.x;
    if (idx >= N_NODES * H) return;
    int c = idx & (H - 1);
    h[idx] = x[idx >> 7] * ew[c] + eb[c];
}

__global__ __launch_bounds__(256) void dist_kernel(
    const float* __restrict__ pos, const int* __restrict__ ei,
    const float* __restrict__ co, const float* __restrict__ uc,
    float* __restrict__ dist)
{
    int e = blockIdx.x * blockDim.x + threadIdx.x;
    if (e >= N_EDGES) return;
    int s = ei[e];
    int d = ei[N_EDGES + e];
    float c0 = co[e * 3 + 0], c1 = co[e * 3 + 1], c2 = co[e * 3 + 2];
    float acc = 0.f;
#pragma unroll
    for (int j = 0; j < 3; ++j) {
        float off = c0 * uc[0 * 3 + j] + c1 * uc[1 * 3 + j] + c2 * uc[2 * 3 + j];
        float diff = pos[(size_t)d * 3 + j] - off - pos[(size_t)s * 3 + j];
        acc += diff * diff;
    }
    dist[e] = sqrtf(acc);
}

__global__ __launch_bounds__(256) void pool_kernel(
    const float* __restrict__ h, const int* __restrict__ bid,
    float* __restrict__ pooled)
{
    int idx = blockIdx.x * blockDim.x + threadIdx.x;
    if (idx >= N_NODES * H) return;
    int n = idx >> 7, c = idx & (H - 1);
    atomicAdd(&pooled[(size_t)bid[n] * H + c], h[idx]);
}

__global__ __launch_bounds__(128) void pred_kernel(
    const float* __restrict__ pooled, const float* __restrict__ w1,
    const float* __restrict__ b1, const float* __restrict__ w2,
    const float* __restrict__ b2, float* __restrict__ out)
{
    __shared__ float part[128];
    int g = blockIdx.x, j = threadIdx.x;
    float s = b1[j];
    for (int k = 0; k < H; ++k) s += pooled[(size_t)g * H + k] * w1[(size_t)k * H + j];
    s = fmaxf(s, 0.f);
    part[j] = s * w2[j];
    __syncthreads();
#pragma unroll
    for (int off = 64; off > 0; off >>= 1) {
        if (j < off) part[j] += part[j + off];
        __syncthreads();
    }
    if (j == 0) out[g] = part[0] + b2[0];
}

// -------------------- fused message kernel (32 edges / block) --------------
__global__ __launch_bounds__(256) void msg_kernel(
    const float* __restrict__ h, const float* __restrict__ dist,
    const int* __restrict__ ei,
    const _Float16* __restrict__ w1p, const float* __restrict__ b1,
    const float* __restrict__ g1, const float* __restrict__ be1,
    const _Float16* __restrict__ w2p, const float* __restrict__ b2,
    const float* __restrict__ g2, const float* __restrict__ be2,
    float* __restrict__ aggr)
{
    __shared__ _Float16 Ain[TM * KM1];
    __shared__ float    St[TM * H];
    __shared__ _Float16 A2[TM * H];
    __shared__ float    red[TM * 8], red2[TM * 8];
    __shared__ float    mu[TM], rs[TM];
    __shared__ int      srcs[TM], dsts[TM];

    int tid  = threadIdx.x;
    int lane = tid & 31;
    int w    = tid >> 5;
    int e0   = blockIdx.x * TM;

    if (tid < TM)       srcs[tid] = ei[e0 + tid];
    else if (tid < 2*TM) dsts[tid - TM] = ei[N_EDGES + e0 + (tid - TM)];
    __syncthreads();

    // Gather: Ain[r][k] = [ h[dst] | h[src] | dist | 0-pad ]
    for (int idx = tid; idx < TM * KM1; idx += 256) {
        int r = idx / KM1, k = idx - r * KM1;
        float v;
        if (k < H)            v = h[(size_t)dsts[r] * H + k];
        else if (k < 2 * H)   v = h[(size_t)srcs[r] * H + (k - H)];
        else if (k == 2 * H)  v = dist[e0 + r];
        else                  v = 0.0f;
        Ain[idx] = (_Float16)v;
    }
    __syncthreads();

    // GEMM1: [32 x 288] x [288 x 128]
    v8f acc0, acc1;
    wmma_tile2(Ain, KM1, KM1 / 32, w1p, w, lane, acc0, acc1);
    store_tiles(St, acc0, acc1, b1, w, lane);
    __syncthreads();

    // LN1 + ReLU -> A2 (f16)
    ln_stats(St, tid, red, red2, mu, rs);
    for (int idx = tid; idx < TM * H; idx += 256) {
        int r = idx >> 7, c = idx & (H - 1);
        float v = (St[idx] - mu[r]) * rs[r] * g1[c] + be1[c];
        A2[idx] = (_Float16)fmaxf(v, 0.f);
    }
    __syncthreads();

    // GEMM2: [32 x 128] x [128 x 128]
    wmma_tile2(A2, H, K2 / 32, w2p, w, lane, acc0, acc1);
    store_tiles(St, acc0, acc1, b2, w, lane);
    __syncthreads();

    // LN2 + ReLU + scatter-add to aggr[dst]
    ln_stats(St, tid, red, red2, mu, rs);
    for (int idx = tid; idx < TM * H; idx += 256) {
        int r = idx >> 7, c = idx & (H - 1);
        float v = (St[idx] - mu[r]) * rs[r] * g2[c] + be2[c];
        v = fmaxf(v, 0.f);
        atomicAdd(&aggr[(size_t)dsts[r] * H + c], v);
    }
}

// -------------------- fused update kernel (32 nodes / block) ---------------
__global__ __launch_bounds__(256) void upd_kernel(
    float* __restrict__ h, const float* __restrict__ aggr,
    const _Float16* __restrict__ w1p, const float* __restrict__ b1,
    const float* __restrict__ g1, const float* __restrict__ be1,
    const _Float16* __restrict__ w2p, const float* __restrict__ b2,
    const float* __restrict__ g2, const float* __restrict__ be2)
{
    __shared__ _Float16 Ain[TM * KU1];
    __shared__ float    St[TM * H];
    __shared__ _Float16 A2[TM * H];
    __shared__ float    red[TM * 8], red2[TM * 8];
    __shared__ float    mu[TM], rs[TM];

    int tid  = threadIdx.x;
    int lane = tid & 31;
    int w    = tid >> 5;
    int n0   = blockIdx.x * TM;
    int nvalid = N_NODES - n0;            // rows valid in this block
    if (nvalid > TM) nvalid = TM;

    // Ain[r][k] = [ h[n] | aggr[n] ]  (zero-fill tail rows)
    for (int idx = tid; idx < TM * KU1; idx += 256) {
        int r = idx >> 8, k = idx & (KU1 - 1);
        float v = 0.0f;
        if (r < nvalid) {
            int n = n0 + r;
            v = (k < H) ? h[(size_t)n * H + k]
                        : aggr[(size_t)n * H + (k - H)];
        }
        Ain[idx] = (_Float16)v;
    }
    __syncthreads();

    v8f acc0, acc1;
    wmma_tile2(Ain, KU1, KU1 / 32, w1p, w, lane, acc0, acc1);
    store_tiles(St, acc0, acc1, b1, w, lane);
    __syncthreads();

    ln_stats(St, tid, red, red2, mu, rs);
    for (int idx = tid; idx < TM * H; idx += 256) {
        int r = idx >> 7, c = idx & (H - 1);
        float v = (St[idx] - mu[r]) * rs[r] * g1[c] + be1[c];
        A2[idx] = (_Float16)fmaxf(v, 0.f);
    }
    __syncthreads();

    wmma_tile2(A2, H, K2 / 32, w2p, w, lane, acc0, acc1);
    store_tiles(St, acc0, acc1, b2, w, lane);
    __syncthreads();

    ln_stats(St, tid, red, red2, mu, rs);
    for (int idx = tid; idx < TM * H; idx += 256) {
        int r = idx >> 7, c = idx & (H - 1);
        if (r < nvalid) {
            float v = (St[idx] - mu[r]) * rs[r] * g2[c] + be2[c];
            v = fmaxf(v, 0.f);
            h[(size_t)(n0 + r) * H + c] += v;   // residual
        }
    }
}

} // anonymous namespace

// ---------------------------------------------------------------------------
extern "C" void kernel_launch(void* const* d_in, const int* in_sizes, int n_in,
                              void* d_out, int out_size, void* d_ws, size_t ws_size,
                              hipStream_t stream)
{
    (void)in_sizes; (void)n_in; (void)out_size; (void)ws_size;

    const float* x       = (const float*)d_in[0];
    const float* pos     = (const float*)d_in[1];
    const int*   ei      = (const int*)  d_in[2];
    const float* cell    = (const float*)d_in[3];
    const float* ucell   = (const float*)d_in[4];
    const int*   bid     = (const int*)  d_in[5];
    const float* emb_w   = (const float*)d_in[6];
    const float* emb_b   = (const float*)d_in[7];
    const float* msg_w1  = (const float*)d_in[8];
    const float* msg_b1  = (const float*)d_in[9];
    const float* msg_g1  = (const float*)d_in[10];
    const float* msg_be1 = (const float*)d_in[11];
    const float* msg_w2  = (const float*)d_in[12];
    const float* msg_b2  = (const float*)d_in[13];
    const float* msg_g2  = (const float*)d_in[14];
    const float* msg_be2 = (const float*)d_in[15];
    const float* upd_w1  = (const float*)d_in[16];
    const float* upd_b1  = (const float*)d_in[17];
    const float* upd_g1  = (const float*)d_in[18];
    const float* upd_be1 = (const float*)d_in[19];
    const float* upd_w2  = (const float*)d_in[20];
    const float* upd_b2  = (const float*)d_in[21];
    const float* upd_g2  = (const float*)d_in[22];
    const float* upd_be2 = (const float*)d_in[23];
    const float* pred_w1 = (const float*)d_in[24];
    const float* pred_b1 = (const float*)d_in[25];
    const float* pred_w2 = (const float*)d_in[26];
    const float* pred_b2 = (const float*)d_in[27];
    float* out = (float*)d_out;

    // ---- workspace carve (256B aligned) ----
    auto align256 = [](size_t s) { return (s + 255) & ~(size_t)255; };
    char* ws = (char*)d_ws;
    float* h      = (float*)ws;  ws += align256((size_t)N_NODES * H * 4);
    float* aggr   = (float*)ws;  ws += align256((size_t)N_NODES * H * 4);
    float* dist   = (float*)ws;  ws += align256((size_t)N_EDGES * 4);
    float* pooled = (float*)ws;  ws += align256((size_t)G * H * 4);
    _Float16* wm1 = (_Float16*)ws; ws += align256((size_t)DEPTH * KM1 * H * 2);
    _Float16* wm2 = (_Float16*)ws; ws += align256((size_t)DEPTH * K2  * H * 2);
    _Float16* wu1 = (_Float16*)ws; ws += align256((size_t)DEPTH * KU1 * H * 2);
    _Float16* wu2 = (_Float16*)ws; ws += align256((size_t)DEPTH * K2  * H * 2);

    // ---- pack weights to fragment-swizzled f16 ----
    for (int l = 0; l < DEPTH; ++l) {
        int kt;
        kt = KM1 / 32;
        pack_w_kernel<<<(kt * 4096 + 255) / 256, 256, 0, stream>>>(
            msg_w1 + (size_t)l * 257 * H, 257, kt, wm1 + (size_t)l * KM1 * H);
        kt = K2 / 32;
        pack_w_kernel<<<(kt * 4096 + 255) / 256, 256, 0, stream>>>(
            msg_w2 + (size_t)l * K2 * H, K2, kt, wm2 + (size_t)l * K2 * H);
        kt = KU1 / 32;
        pack_w_kernel<<<(kt * 4096 + 255) / 256, 256, 0, stream>>>(
            upd_w1 + (size_t)l * KU1 * H, KU1, kt, wu1 + (size_t)l * KU1 * H);
        kt = K2 / 32;
        pack_w_kernel<<<(kt * 4096 + 255) / 256, 256, 0, stream>>>(
            upd_w2 + (size_t)l * K2 * H, K2, kt, wu2 + (size_t)l * K2 * H);
    }

    // ---- embedding + edge distances ----
    emb_kernel<<<(N_NODES * H + 255) / 256, 256, 0, stream>>>(x, emb_w, emb_b, h);
    dist_kernel<<<(N_EDGES + 255) / 256, 256, 0, stream>>>(pos, ei, cell, ucell, dist);

    // ---- message passing layers ----
    for (int l = 0; l < DEPTH; ++l) {
        hipMemsetAsync(aggr, 0, (size_t)N_NODES * H * sizeof(float), stream);
        msg_kernel<<<N_EDGES / TM, 256, 0, stream>>>(
            h, dist, ei,
            wm1 + (size_t)l * KM1 * H, msg_b1 + (size_t)l * H,
            msg_g1 + (size_t)l * H,    msg_be1 + (size_t)l * H,
            wm2 + (size_t)l * K2 * H,  msg_b2 + (size_t)l * H,
            msg_g2 + (size_t)l * H,    msg_be2 + (size_t)l * H,
            aggr);
        upd_kernel<<<(N_NODES + TM - 1) / TM, 256, 0, stream>>>(
            h, aggr,
            wu1 + (size_t)l * KU1 * H, upd_b1 + (size_t)l * H,
            upd_g1 + (size_t)l * H,    upd_be1 + (size_t)l * H,
            wu2 + (size_t)l * K2 * H,  upd_b2 + (size_t)l * H,
            upd_g2 + (size_t)l * H,    upd_be2 + (size_t)l * H);
    }

    // ---- pooling + prediction head ----
    hipMemsetAsync(pooled, 0, (size_t)G * H * sizeof(float), stream);
    pool_kernel<<<(N_NODES * H + 255) / 256, 256, 0, stream>>>(h, bid, pooled);
    pred_kernel<<<G, 128, 0, stream>>>(pooled, pred_w1, pred_b1, pred_w2, pred_b2, out);
}
